// BatchedACE_42262478193315
// MI455X (gfx1250) — matrix operand order repaired and checked by reading, not compiled, not run on hardware
//
#include <hip/hip_runtime.h>

// Problem constants (match reference)
#define D_KD 64
#define K_BITS 4
#define L_TABLES 8
#define R_CORNERS 16
#define B_SZ 2
#define T_LEN 512
#define H_HEADS 8
#define M_ENS 2
#define N_SEQ (M_ENS * B_SZ * H_HEADS) /* 32 */
#define TC 32                          /* timestep chunk */
#define NCHUNK (T_LEN / TC)            /* 16 */
#define LR 128                         /* L_TABLES * R_CORNERS */

typedef __bf16 bf16;
typedef bf16 v16bf __attribute__((ext_vector_type(16)));
typedef float v8f __attribute__((ext_vector_type(8)));
typedef float f32x4 __attribute__((ext_vector_type(4)));
typedef unsigned int u32x4 __attribute__((ext_vector_type(4)));

union Frag {
    v16bf v;
    u32x4 q[2];
    bf16 h[16];
};

union Pack4 {
    bf16 h[4];
    unsigned long long u;
};

__device__ inline v8f zero8() {
    v8f z;
#pragma unroll
    for (int i = 0; i < 8; ++i) z[i] = 0.0f;
    return z;
}

// Branch-free tanh: prefer native gfx1250 V_TANH_F32 if clang exposes it.
__device__ inline float tanh_fast(float x) {
#if __has_builtin(__builtin_amdgcn_tanhf)
    return __builtin_amdgcn_tanhf(x);
#elif __has_builtin(__builtin_amdgcn_tanh_f32)
    return __builtin_amdgcn_tanh_f32(x);
#else
    // tanh(x) = 1 - 2/(exp2(2*log2e*x) + 1); exp2 saturates cleanly at both ends.
    const float e = __builtin_amdgcn_exp2f(2.885390082f * x);
    return __builtin_fmaf(-2.0f, __builtin_amdgcn_rcpf(e + 1.0f), 1.0f);
#endif
}

// A-fragment: 16x32 bf16 tile, row-major LDS. Lane L holds row M=L&15;
// element j -> K = (L>>4)*16 + j  ==> one contiguous 32B run per lane.
__device__ inline v16bf load_a_rm(const bf16* base, int m0, int k0, int lda, int lane) {
    const bf16* p = base + (m0 + (lane & 15)) * lda + k0 + ((lane >> 4) & 1) * 16;
    Frag f;
    f.q[0] = *(const u32x4*)(p);
    f.q[1] = *(const u32x4*)(p + 8);
    return f.v;
}

// B-fragment: 32x16 (KxN) tile read from TRANSPOSED storage Bt[N][K] (row-major,
// leading dim ldt). Lane L holds column N=L&15; element j -> K = (L>>4)*16 + j
// ==> one contiguous 32B run per lane.
__device__ inline v16bf load_b_t(const bf16* baseT, int k0, int n0, int ldt, int lane) {
    const bf16* p = baseT + (n0 + (lane & 15)) * ldt + k0 + ((lane >> 4) & 1) * 16;
    Frag f;
    f.q[0] = *(const u32x4*)(p);
    f.q[1] = *(const u32x4*)(p + 8);
    return f.v;
}

// B-fragment from row-major KxN storage (column-strided; only used twice per
// wave per chunk, for the P-matmul operand held lr-major for the S-update A).
__device__ inline v16bf load_b_rm(const bf16* base, int k0, int n0, int ldb, int lane) {
    const bf16* p = base + (k0 + ((lane >> 4) & 1) * 16) * ldb + n0 + (lane & 15);
    Frag f;
#pragma unroll
    for (int j = 0; j < 16; ++j) f.h[j] = p[j * ldb];
    return f.v;
}

__device__ inline v8f wmma_bf16(v16bf a, v16bf b, v8f c) {
    return __builtin_amdgcn_wmma_f32_16x16x32_bf16(false, a, false, b, (short)0, c, false, false);
}

// LDS layout (bytes), total 57344, with phase-safe aliasing:
//   [    0,  4096) sPlanesT bf16 [32][64]  (planes transposed: [lk][d])
//   [ 4096,  8192) sK bf16 [32][64]        -- alias sP f32 [32][32] after phase 1
//   [ 8192, 12288) sQ bf16 [32][64]        -- alias sPm bf16 [32][32] after phase 1
//   [12288, 16384) sVt bf16 [64][32]       (V transposed: [d][s])
//   [16384, 20480) hK f32 [32][32]
//   [20480, 24576) hQ f32 [32][32]
//   [24576, 32768) sPKT bf16 [128][32]     (probsK, lr-major)
//   [32768, 40960) sW bf16 [32][128]       -- alias sO f32 [32][64] after phase 3
//   [40960, 57344) sSt bf16 [64][128]      (S_prev transposed: [d][lr])
__global__ __launch_bounds__(128) void ace_linear_attn_kernel(
    const float* __restrict__ Khf, const float* __restrict__ Vhf,
    const float* __restrict__ Qhf, const float* __restrict__ planesT,
    float* __restrict__ out) {
    __shared__ __align__(16) char smem[57344];
    bf16* sPlanesT = (bf16*)(smem + 0);
    bf16* sK = (bf16*)(smem + 4096);
    bf16* sQ = (bf16*)(smem + 8192);
    bf16* sVt = (bf16*)(smem + 12288);
    float* hK = (float*)(smem + 16384);
    float* hQ = (float*)(smem + 20480);
    bf16* sPKT = (bf16*)(smem + 24576);
    bf16* sW = (bf16*)(smem + 32768);
    bf16* sSt = (bf16*)(smem + 40960);
    float* sP = (float*)(smem + 4096);   // alias sK
    bf16* sPm = (bf16*)(smem + 8192);    // alias sQ
    float* sO = (float*)(smem + 32768);  // alias sW

    const int tid = threadIdx.x;
    const int lane = tid & 31;
    const int w = tid >> 5; // wave id; wave w owns tables {2w,2w+1} = lr rows [32w,32w+32)
    const int n = blockIdx.x;
    const int h = n % H_HEADS;
    const int mb = n / H_HEADS;
    const long long base0 = (long long)mb * T_LEN * H_HEADS * D_KD + (long long)h * D_KD;
    const int strideT = H_HEADS * D_KD; // 512

    // Hyperplanes once, transposed: sPlanesT[lk][d] = planes_T[d][lk]
    for (int i = tid; i < D_KD * L_TABLES * K_BITS; i += 128) {
        const int d = i >> 5, lk = i & 31;
        sPlanesT[lk * D_KD + d] = (bf16)planesT[i];
    }

    // Per-wave state tiles: S_w is 32(lr) x 64(d), as 2x4 WMMA f32 C-tiles.
    v8f Stile[2][4];
    v8f Otile[2][4];
#pragma unroll
    for (int rt = 0; rt < 2; ++rt)
#pragma unroll
        for (int ct = 0; ct < 4; ++ct) Stile[rt][ct] = zero8();

    float A_run = 0.0f;
    const int r = lane & 15;
    const int l = 2 * w + (lane >> 4);
    const int lr = 32 * w + lane;
    // softmax over corners factorizes: p_r = prod_k sigmoid(2*s_kr*g_k)
    //   = prod_k rcp(1 + exp2(sgn2_k * g_k)),  sgn2_k = -2*log2e*s_kr
    float sgn2[K_BITS];
#pragma unroll
    for (int k = 0; k < K_BITS; ++k)
        sgn2[k] = ((r >> (3 - k)) & 1) ? -2.885390082f : 2.885390082f;

    __syncthreads();

    for (int c = 0; c < NCHUNK; ++c) {
        const int t0 = c * TC;
        const int half = lane >> 4;
        const int ncol = lane & 15;

        // ---- Phase 0: spill S_prev transposed (bf16); stage K/Q/V chunk; prefetch ----
        {
#pragma unroll
            for (int rt = 0; rt < 2; ++rt)
#pragma unroll
                for (int ct = 0; ct < 4; ++ct) {
#pragma unroll
                    for (int v = 0; v < 8; ++v)
                        sSt[(16 * ct + ncol) * LR + 32 * w + 16 * rt + v + 8 * half] =
                            (bf16)Stile[rt][ct][v];
                    Otile[rt][ct] = zero8();
                }
#pragma unroll
            for (int i = 0; i < 4; ++i) {
                const int idx4 = i * 128 + tid; // 512 float4s = 32x64
                const int t = idx4 >> 4;
                const int d = (idx4 & 15) * 4;
                const long long g = base0 + (long long)(t0 + t) * strideT + d;
                const f32x4 kv = *(const f32x4*)(Khf + g);
                const f32x4 qv = *(const f32x4*)(Qhf + g);
                const f32x4 vv = *(const f32x4*)(Vhf + g);
                Pack4 pk, pq;
#pragma unroll
                for (int e = 0; e < 4; ++e) {
                    pk.h[e] = (bf16)kv[e];
                    pq.h[e] = (bf16)qv[e];
                    sVt[(d + e) * TC + t] = (bf16)vv[e]; // transpose V
                }
                *(unsigned long long*)(sK + t * D_KD + d) = pk.u;
                *(unsigned long long*)(sQ + t * D_KD + d) = pq.u;
                if (c + 1 < NCHUNK) { // hide next chunk's HBM latency under compute
                    const long long gn = g + (long long)TC * strideT;
                    __builtin_prefetch(Khf + gn, 0, 3);
                    __builtin_prefetch(Qhf + gn, 0, 3);
                    __builtin_prefetch(Vhf + gn, 0, 3);
                }
            }
        }
        __syncthreads();

        // ---- Phase 1: proj = X(32x64) @ planes(64x32) via WMMA; h = tanh(proj)/8 ----
        {
            const bf16* src = (w < 2) ? sK : sQ;
            float* dst = (w < 2) ? hK : hQ;
            const int rt = w & 1;
#pragma unroll
            for (int ct = 0; ct < 2; ++ct) {
                v8f acc = zero8();
#pragma unroll
                for (int ks = 0; ks < 2; ++ks) {
                    v16bf a = load_a_rm(src, 16 * rt, 32 * ks, D_KD, lane);
                    v16bf b = load_b_t(sPlanesT, 32 * ks, 16 * ct, D_KD, lane);
                    acc = wmma_bf16(a, b, acc);
                }
#pragma unroll
                for (int v = 0; v < 8; ++v) {
                    const int t = 16 * rt + v + 8 * half;
                    dst[t * 32 + 16 * ct + ncol] = tanh_fast(acc[v]) * 0.125f;
                }
            }
        }
        __syncthreads();

        // ---- Phase 2: factorized corner-softmax, running A, weights ----
        {
            for (int t = 0; t < TC; ++t) {
                const f32x4 gk = *(const f32x4*)(hK + t * 32 + l * K_BITS);
                const f32x4 gq = *(const f32x4*)(hQ + t * 32 + l * K_BITS);
                float pk = 1.0f, pq = 1.0f;
#pragma unroll
                for (int k = 0; k < K_BITS; ++k) {
                    pk *= __builtin_amdgcn_rcpf(1.0f + __builtin_amdgcn_exp2f(sgn2[k] * gk[k]));
                    pq *= __builtin_amdgcn_rcpf(1.0f + __builtin_amdgcn_exp2f(sgn2[k] * gq[k]));
                }
                A_run += pk;
                const float wt = pq * __builtin_amdgcn_rcpf(A_run + 1e-6f);
                sPKT[lr * TC + t] = (bf16)pk;
                sW[t * LR + lr] = (bf16)wt;
            }
            for (int i = tid; i < TC * TC; i += 128) sP[i] = 0.0f;
        }
        __syncthreads();

        // ---- Phase 3: O1 = W_w @ S_prev_w (regs); P_partial = W_w @ PK_w^T ----
        {
#pragma unroll
            for (int rt = 0; rt < 2; ++rt) {
                v16bf aW = load_a_rm(sW, 16 * rt, 32 * w, LR, lane);
#pragma unroll
                for (int ct = 0; ct < 4; ++ct) {
                    v16bf b = load_b_t(sSt, 32 * w, 16 * ct, LR, lane);
                    Otile[rt][ct] = wmma_bf16(aW, b, Otile[rt][ct]);
                }
#pragma unroll
                for (int ctp = 0; ctp < 2; ++ctp) {
                    v16bf b = load_b_rm(sPKT, 32 * w, 16 * ctp, TC, lane);
                    v8f p = wmma_bf16(aW, b, zero8());
#pragma unroll
                    for (int v = 0; v < 8; ++v)
                        atomicAdd(&sP[(16 * rt + v + 8 * half) * TC + 16 * ctp + ncol], p[v]);
                }
            }
        }
        __syncthreads();

        // ---- Phase 4: causal mask (s<=t) -> bf16; zero output accumulator ----
        {
            for (int i = tid; i < TC * TC; i += 128) {
                const int t = i >> 5, s = i & 31;
                sPm[i] = (bf16)((s <= t) ? sP[i] : 0.0f);
            }
            for (int i = tid; i < TC * D_KD; i += 128) sO[i] = 0.0f;
        }
        __syncthreads();

        // ---- Phase 5: O2 = P_masked @ V; S_w += PK_w^T @ V; reduce O via ds_add ----
        {
#pragma unroll
            for (int rt = 0; rt < 2; ++rt) {
                v16bf a = load_a_rm(sPm, 16 * rt, 0, TC, lane);
                v16bf b = load_b_t(sVt, 0, 16 * w, TC, lane);
                v8f o2 = wmma_bf16(a, b, zero8());
#pragma unroll
                for (int v = 0; v < 8; ++v)
                    atomicAdd(&sO[(16 * rt + v + 8 * half) * D_KD + 16 * w + ncol], o2[v]);
            }
#pragma unroll
            for (int rt = 0; rt < 2; ++rt) {
                v16bf a = load_a_rm(sPKT, 32 * w + 16 * rt, 0, TC, lane);
#pragma unroll
                for (int ct = 0; ct < 4; ++ct) {
                    v16bf b = load_b_t(sVt, 0, 16 * ct, TC, lane);
                    Stile[rt][ct] = wmma_bf16(a, b, Stile[rt][ct]);
                }
            }
#pragma unroll
            for (int rt = 0; rt < 2; ++rt)
#pragma unroll
                for (int ct = 0; ct < 4; ++ct)
#pragma unroll
                    for (int v = 0; v < 8; ++v)
                        atomicAdd(&sO[(16 * rt + v + 8 * half) * D_KD + 16 * ct + ncol],
                                  Otile[rt][ct][v]);
        }
        __syncthreads();

        // ---- Phase 6: store chunk output (vectorized) ----
#pragma unroll
        for (int i = 0; i < 4; ++i) {
            const int idx4 = i * 128 + tid;
            const int t = idx4 >> 4;
            const int d = (idx4 & 15) * 4;
            const f32x4 o = *(const f32x4*)(sO + t * D_KD + d);
            *(f32x4*)(out + base0 + (long long)(t0 + t) * strideT + d) = o;
        }
        __syncthreads();
    }
}

extern "C" void kernel_launch(void* const* d_in, const int* in_sizes, int n_in,
                              void* d_out, int out_size, void* d_ws, size_t ws_size,
                              hipStream_t stream) {
    const float* Khf = (const float*)d_in[0];
    const float* Vhf = (const float*)d_in[1];
    const float* Qhf = (const float*)d_in[2];
    const float* planesT = (const float*)d_in[3];
    // d_in[4] (protos_T) not needed: corner signs derived analytically.
    float* out = (float*)d_out;
    (void)in_sizes; (void)n_in; (void)out_size; (void)d_ws; (void)ws_size;
    ace_linear_attn_kernel<<<dim3(N_SEQ), dim3(128), 0, stream>>>(Khf, Vhf, Qhf, planesT, out);
}
